// Rec1_43748536877643
// MI455X (gfx1250) — compile-verified, matrix operand order checked by reading, not run on hardware
//
#include <hip/hip_runtime.h>

#define DIM   256
#define HID   4096
#define LSEQ  2048
#define NBAT  4
#define PITCH 264   // DIM + 8 pad, multiple of 8 -> 16B aligned rows

typedef __attribute__((ext_vector_type(8)))  __bf16 v8bf;
typedef __attribute__((ext_vector_type(16))) __bf16 v16bf;
typedef __attribute__((ext_vector_type(8)))  float  v8f;
typedef __attribute__((ext_vector_type(4)))  unsigned int u32x4;
typedef __attribute__((ext_vector_type(8)))  int i32x8;
typedef __attribute__((ext_vector_type(4)))  int i32x4;

// TDM builtin exists on both toolchains but with different arity:
//   ROCm 7.2 (clang-22): 5 args (g0 x4, g1 x8, g2 x4, g3 x4, cpol)
//   amdgpu-toolchain (clang-23): 6 args (g0 x4, g1 x8, x4, x4, x8, cpol)
#if __has_builtin(__builtin_amdgcn_tensor_load_to_lds)
#define USE_TDM 1
#else
#define USE_TDM 0
#endif

// Build a 16x32 bf16 WMMA fragment (A: MxK, or B mirrored KxN) from an LDS row.
// Per ISA 7.12.2: lanes 0-15 cover K = {kk*32+0..7, kk*32+16..23},
// lanes 16-31 cover K = {kk*32+8..15, kk*32+24..31}. Two 16-byte contiguous runs.
__device__ __forceinline__ v16bf load_frag(const __bf16* rowbase, int kk, int lane) {
    int base = (lane & 16) ? 8 : 0;
    v8bf lo = *(const v8bf*)(rowbase + kk * 32 + base);
    v8bf hi = *(const v8bf*)(rowbase + kk * 32 + base + 16);
    return __builtin_shufflevector(lo, hi, 0,1,2,3,4,5,6,7,8,9,10,11,12,13,14,15);
}

#if USE_TDM
// TDM: DMA one 16x256 bf16 tile (row stride 256 elems) from global into LDS,
// with HW padding of 4 DWORDs every 128 DWORDs -> effective LDS pitch = PITCH.
// D# layout per ISA 08_async_tensor §8.3/§8.4. lds_addr = flat addr truncated
// to 32 bits (aperture rule: addr[31:0] is the wave-relative LDS offset).
__device__ __forceinline__ void tdm_load_x_chunk(const __bf16* gsrc, unsigned int lds_addr) {
    unsigned long long ga = (unsigned long long)gsrc;
    u32x4 g0;
    g0[0] = 1u;                                              // count=1, user desc
    g0[1] = lds_addr;                                        // LDS byte address
    g0[2] = (unsigned int)ga;                                // global_addr[31:0]
    g0[3] = ((unsigned int)(ga >> 32) & 0x01FFFFFFu) | (2u << 30); // addr[56:32] | type=2
    i32x8 g1;
    g1[0] = (int)((1u << 16)    // data_size = 2 bytes
                | (1u << 20)    // pad_enable
                | (6u << 22)    // pad_interval: 128 DWORDs (one 256-bf16 row)
                | (3u << 25));  // pad_amount: 4 DWORDs (8 bf16) -> pitch 264
    g1[1] = (int)(256u << 16);  // tensor_dim0 = 256 (low16; abar addr = 0)
    g1[2] = (int)(16u  << 16);  // tensor_dim0 hi = 0 | tensor_dim1 = 16
    g1[3] = (int)(256u << 16);  // tensor_dim1 hi = 0 | tile_dim0 = 256
    g1[4] = 16;                 // tile_dim1 = 16, tile_dim2 = 0 (2D)
    g1[5] = 256;                // tensor_dim0_stride = 256 elems (low32)
    g1[6] = 0;                  // stride0 hi | tensor_dim1_stride lo (unused, 2D)
    g1[7] = 0;
    i32x4 z4 = {0, 0, 0, 0};
#if defined(__clang_major__) && (__clang_major__ >= 23)
    i32x8 z8 = {0, 0, 0, 0, 0, 0, 0, 0};
    __builtin_amdgcn_tensor_load_to_lds(g0, g1, z4, z4, z8, 0);   // 6-arg form
#else
    __builtin_amdgcn_tensor_load_to_lds(g0, g1, z4, z4, 0);       // 5-arg form
#endif
}
#endif

// ---------- converters ----------
__global__ void cvt_bf16(const float* __restrict__ in, __bf16* __restrict__ out, int n) {
    int i = blockIdx.x * blockDim.x + threadIdx.x;
    if (i < n) out[i] = (__bf16)in[i];
}

// WC [DIM, HID] f32  ->  WCT [HID, DIM] bf16
__global__ void cvt_wct(const float* __restrict__ wc, __bf16* __restrict__ wct) {
    int i = blockIdx.x * blockDim.x + threadIdx.x;
    if (i < HID * DIM) {
        int h = i / DIM, d = i % DIM;
        wct[i] = (__bf16)wc[d * HID + h];
    }
}

// ---------- fused GEMM1 (b = x @ WB^T + bB) + linear scan ----------
// grid: NBAT * (HID/128) blocks, 256 threads (8 waves, 16 channels each).
// x chunks double-buffered; TDM DMA for chunk c+1 overlaps WMMA on chunk c.
__global__ __launch_bounds__(256)
void gemm1_scan(const __bf16* __restrict__ xbf, const __bf16* __restrict__ wbbf,
                const float* __restrict__ bB, const float* __restrict__ A,
                __bf16* __restrict__ hbuf) {
    __shared__ __align__(16) __bf16 wb_lds[128 * PITCH];
    __shared__ __align__(16) __bf16 x_lds[2][16 * PITCH];

    const int batch = blockIdx.x >> 5;          // /32
    const int h0    = (blockIdx.x & 31) * 128;
    const int tid   = threadIdx.x;
    const int wave  = tid >> 5;
    const int lane  = tid & 31;
    const int chan  = h0 + wave * 16 + (lane & 15);

#if USE_TDM
    if (wave == 0)      // prime the pipeline: DMA chunk 0 while WB tile stages
        tdm_load_x_chunk(&xbf[(size_t)batch * LSEQ * DIM],
                         (unsigned int)(unsigned long long)&x_lds[0][0]);
#endif

    // stage WB tile (128 rows x 256 cols, bf16) once
    #pragma unroll
    for (int i = 0; i < 16; ++i) {
        int v = tid + i * 256;                  // 4096 v8 loads
        int row = v >> 5, c8 = (v & 31) * 8;
        *(v8bf*)&wb_lds[row * PITCH + c8] =
            *(const v8bf*)&wbbf[(size_t)(h0 + row) * DIM + c8];
    }

    const float av   = A[chan];
    const float a    = 1.0f / (1.0f + __expf(-av));   // sigmoid decay
    const float bias = bB[chan];
    const float a2 = a * a, a4 = a2 * a2, a8 = a4 * a4;
    float carry = 0.0f;                               // h_{-1} = 0

    for (int l0 = 0; l0 < LSEQ; l0 += 16) {
        const int cur = (l0 >> 4) & 1;
#if USE_TDM
        if (wave == 0) __builtin_amdgcn_s_wait_tensorcnt(0);
        __syncthreads();    // chunk `cur` ready; everyone done reading cur^1
        if (wave == 0 && l0 + 16 < LSEQ)    // DMA next chunk behind the WMMAs
            tdm_load_x_chunk(&xbf[((size_t)batch * LSEQ + l0 + 16) * DIM],
                             (unsigned int)(unsigned long long)&x_lds[cur ^ 1][0]);
#else
        __syncthreads();
        #pragma unroll
        for (int i = 0; i < 2; ++i) {           // coop stage 16x256 bf16
            int v = tid + i * 256;
            int row = v >> 5, c8 = (v & 31) * 8;
            *(v8bf*)&x_lds[cur][row * PITCH + c8] =
                *(const v8bf*)&xbf[((size_t)batch * LSEQ + l0 + row) * DIM + c8];
        }
        if (l0 + 16 < LSEQ)
            __builtin_prefetch(
                &xbf[((size_t)batch * LSEQ + l0 + 16 + (tid >> 4)) * DIM + (tid & 15) * 16], 0, 0);
        __syncthreads();
#endif

        // b_chunk[16t x 16h] = x_chunk @ WB_tile^T  (K = 256 in 8 WMMA steps)
        v8f acc = {0.f,0.f,0.f,0.f,0.f,0.f,0.f,0.f};
        const __bf16* arow = &x_lds[cur][(lane & 15) * PITCH];
        const __bf16* brow = &wb_lds[(wave * 16 + (lane & 15)) * PITCH];
        #pragma unroll
        for (int kk = 0; kk < 8; ++kk) {
            v16bf af = load_frag(arow, kk, lane);
            v16bf bf = load_frag(brow, kk, lane);
            acc = __builtin_amdgcn_wmma_f32_16x16x32_bf16(false, af, false, bf,
                                                          (short)0, acc, false, false);
        }

        // lane holds channel n = lane&15; rows t = r + (lane>=16 ? 8 : 0)
        float bv[8];
        #pragma unroll
        for (int r = 0; r < 8; ++r) bv[r] = acc[r] + bias;

        // pass 1: local inclusive scan with zero init -> only last value needed
        float loc = 0.0f;
        #pragma unroll
        for (int r = 0; r < 8; ++r) loc = a * loc + bv[r];
        // lower half's h_7 feeds upper half's start
        float l7low = __shfl(loc, lane & 15, 32);
        float h7    = a8 * carry + l7low;
        float start = (lane < 16) ? carry : h7;
        // pass 2: true scan
        float h = start;
        #pragma unroll
        for (int r = 0; r < 8; ++r) { h = a * h + bv[r]; bv[r] = h; }
        // new carry = h_15 (upper half, r = 7), broadcast to both halves
        carry = __shfl(h, (lane & 15) + 16, 32);

        // store h chunk bf16 -> hbuf[batch, l0+t, chan]
        size_t bi = ((size_t)batch * LSEQ + l0 + ((lane & 16) ? 8 : 0)) * HID + chan;
        #pragma unroll
        for (int r = 0; r < 8; ++r) hbuf[bi + (size_t)r * HID] = (__bf16)bv[r];
    }
}

// ---------- GEMM2: out = h @ WC^T + bC ----------
// grid (128, 4): 64-row x 64-col tiles; 8 waves = 2x4; each wave owns two
// 16x16 M sub-tiles (B fragment reused) -> 4 WMMAs per wave per barrier pair.
#define KC  64
#define AP2 72   // KC + 8, multiple of 8 -> 16B-aligned rows
#define BP2 72
__global__ __launch_bounds__(256)
void gemm2(const __bf16* __restrict__ hbuf, const __bf16* __restrict__ wct,
           const float* __restrict__ bC, float* __restrict__ out) {
    __shared__ __align__(16) __bf16 a_lds[64 * AP2];   // 64 M-rows x 64 K
    __shared__ __align__(16) __bf16 b_lds[64 * BP2];   // 64 N-rows x 64 K (transposed)

    const int m0 = blockIdx.x * 64;
    const int d0 = blockIdx.y * 64;
    const int tid = threadIdx.x, wave = tid >> 5, lane = tid & 31;
    const int mi = wave >> 2;   // 0..1 -> M rows mi*32 + {0,16}
    const int ni = wave & 3;    // 0..3 -> N sub-tile

    v8f acc0 = {0.f,0.f,0.f,0.f,0.f,0.f,0.f,0.f};
    v8f acc1 = {0.f,0.f,0.f,0.f,0.f,0.f,0.f,0.f};

    for (int k0 = 0; k0 < HID; k0 += KC) {
        __syncthreads();
        // A tile: 64 rows x 64 k (row-major in k) = 512 v8 loads
        #pragma unroll
        for (int i = 0; i < 2; ++i) {
            int v = tid + i * 256;
            int row = v >> 3, c8 = (v & 7) * 8;
            *(v8bf*)&a_lds[row * AP2 + c8] =
                *(const v8bf*)&hbuf[(size_t)(m0 + row) * HID + k0 + c8];
        }
        // B tile: WCT[k0+k][d0+n] (coalesced over d) -> b_lds[n][k]
        #pragma unroll
        for (int i = 0; i < 2; ++i) {
            int v = tid + i * 256;
            int k = v >> 3, n8 = (v & 7) * 8;
            v8bf t = *(const v8bf*)&wct[(size_t)(k0 + k) * DIM + d0 + n8];
            #pragma unroll
            for (int j = 0; j < 8; ++j) b_lds[(n8 + j) * BP2 + k] = t[j];
        }
        __syncthreads();

        const __bf16* a0 = &a_lds[(mi * 32 +      (lane & 15)) * AP2];
        const __bf16* a1 = &a_lds[(mi * 32 + 16 + (lane & 15)) * AP2];
        const __bf16* br = &b_lds[(ni * 16 +      (lane & 15)) * BP2];
        #pragma unroll
        for (int kk = 0; kk < 2; ++kk) {
            v16bf bfrag = load_frag(br, kk, lane);
            v16bf af0   = load_frag(a0, kk, lane);
            v16bf af1   = load_frag(a1, kk, lane);
            acc0 = __builtin_amdgcn_wmma_f32_16x16x32_bf16(false, af0, false, bfrag,
                                                           (short)0, acc0, false, false);
            acc1 = __builtin_amdgcn_wmma_f32_16x16x32_bf16(false, af1, false, bfrag,
                                                           (short)0, acc1, false, false);
        }
    }

    const int d    = d0 + ni * 16 + (lane & 15);
    const float bc = bC[d];
    const int mb0  = m0 + mi * 32 + ((lane & 16) ? 8 : 0);
    #pragma unroll
    for (int r = 0; r < 8; ++r) {
        __builtin_nontemporal_store(acc0[r] + bc, &out[(size_t)(mb0 + r)      * DIM + d]);
        __builtin_nontemporal_store(acc1[r] + bc, &out[(size_t)(mb0 + 16 + r) * DIM + d]);
    }
}

extern "C" void kernel_launch(void* const* d_in, const int* in_sizes, int n_in,
                              void* d_out, int out_size, void* d_ws, size_t ws_size,
                              hipStream_t stream) {
    const float* x  = (const float*)d_in[0];   // [4, 2048, 256]
    const float* WB = (const float*)d_in[1];   // [4096, 256]
    const float* bB = (const float*)d_in[2];   // [4096]
    const float* WC = (const float*)d_in[3];   // [256, 4096]
    const float* bC = (const float*)d_in[4];   // [256]
    const float* A  = (const float*)d_in[5];   // [4096]
    float* out = (float*)d_out;                // [4, 2048, 256]

    const size_t nx  = (size_t)NBAT * LSEQ * DIM;   // 2,097,152
    const size_t nwb = (size_t)HID * DIM;           // 1,048,576

    char* ws = (char*)d_ws;
    __bf16* xbf  = (__bf16*)ws;                       ws += nx  * sizeof(__bf16);
    __bf16* wbbf = (__bf16*)ws;                       ws += nwb * sizeof(__bf16);
    __bf16* wct  = (__bf16*)ws;                       ws += nwb * sizeof(__bf16);
    __bf16* hbuf = (__bf16*)ws;                       // 64 MB

    cvt_bf16<<<(int)((nx  + 255) / 256), 256, 0, stream>>>(x,  xbf,  (int)nx);
    cvt_bf16<<<(int)((nwb + 255) / 256), 256, 0, stream>>>(WB, wbbf, (int)nwb);
    cvt_wct <<<(int)((nwb + 255) / 256), 256, 0, stream>>>(WC, wct);

    gemm1_scan<<<NBAT * (HID / 128), 256, 0, stream>>>(xbf, wbbf, bB, A, hbuf);

    dim3 g2(NBAT * LSEQ / 64, DIM / 64);
    gemm2<<<g2, 256, 0, stream>>>(hbuf, wct, bC, out);
}
